// MultiHeadAttention_58531814310501
// MI455X (gfx1250) — matrix operand order repaired and checked by reading, not compiled, not run on hardware
//
#include <hip/hip_runtime.h>
#include <hip/hip_bf16.h>

typedef __attribute__((ext_vector_type(16))) __bf16    v16bf;
typedef __attribute__((ext_vector_type(8)))  float     v8f;
typedef __attribute__((ext_vector_type(4)))  unsigned  u32x4;
typedef __attribute__((ext_vector_type(8)))  int       i32x8;
typedef __attribute__((ext_vector_type(4)))  int       i32x4;

#define D_MODEL 2048
#define SEQ     2048
#define BATCH   2
#define NHEADS  16
#define DHEAD   128
#define MROWS   (BATCH * SEQ)   // 4096

// ---------------------------------------------------------------------------
__device__ __forceinline__ unsigned short f2bf(float f) {
    unsigned u = __float_as_uint(f);
    unsigned r = u + 0x7FFFu + ((u >> 16) & 1u);
    return (unsigned short)(r >> 16);
}

union FragBF { v16bf v; uint4 q[2]; };

// A-fragment 16x32 (MxK) bf16 from row-major src (ld elems/row); LDS or global.
__device__ __forceinline__ v16bf load_frag_a(const unsigned short* src, int ld,
                                             int row0, int k0, int lane) {
    int m = lane & 15, half = lane >> 4;
    const unsigned short* r = src + (size_t)(row0 + m) * ld + k0 + half * 8;
    FragBF f;
    f.q[0] = *(const uint4*)(r);
    f.q[1] = *(const uint4*)(r + 16);
    return f.v;
}

// B-fragment 32x16 (KxN) bf16, storage is B^T row-major: bt[n][k], ld elems/row.
__device__ __forceinline__ v16bf load_frag_b(const unsigned short* bt, int ld,
                                             int n0, int k0, int lane) {
    int n = lane & 15, half = lane >> 4;
    const unsigned short* r = bt + (size_t)(n0 + n) * ld + k0 + half * 8;
    FragBF f;
    f.q[0] = *(const uint4*)(r);
    f.q[1] = *(const uint4*)(r + 16);
    return f.v;
}

__device__ __forceinline__ v8f wmma_bf16(v16bf a, v16bf b, v8f c) {
    return __builtin_amdgcn_wmma_f32_16x16x32_bf16(false, a, false, b,
                                                   (short)0, c, false, false);
}

// Per-lane async DMA: global -> LDS, 16B, tracked by ASYNCcnt.
__device__ __forceinline__ void async_b128(const void* gaddr, void* lds_ptr) {
    unsigned lds = (unsigned)(uintptr_t)lds_ptr;           // low 32b = LDS offset
    unsigned long long g = (unsigned long long)(uintptr_t)gaddr;
    asm volatile("global_load_async_to_lds_b128 %0, %1, off"
                 :: "v"(lds), "v"(g) : "memory");
}
__device__ __forceinline__ void wait_async0() {
    asm volatile("s_wait_asynccnt 0x0" ::: "memory");
}

// ---------------------------------------------------------------------------
// Tensor Data Mover: one-instruction 2D tile DMA global -> LDS with LDS padding.
// tile = tile_rows x tile_k (bf16 elems), source row stride = row_stride elems.
// pad_interval/pad_amount per ISA D# group1: pad after 8<<interval bytes,
// pad amount = (pad_amount+1) DWORDs.
__device__ __forceinline__ void tdm_load_2d(const void* gaddr, void* lds_ptr,
                                            unsigned tile_k, unsigned tile_rows,
                                            unsigned row_stride,
                                            unsigned pad_interval, unsigned pad_amount) {
    unsigned long long ga = (unsigned long long)(uintptr_t)gaddr;
    u32x4 g0;
    g0.x = 1u;                                          // count=1, user mode
    g0.y = (unsigned)(uintptr_t)lds_ptr;                // lds_addr
    g0.z = (unsigned)ga;                                // global_addr[31:0]
    g0.w = ((unsigned)(ga >> 32) & 0x01FFFFFFu) | (2u << 30);  // [56:32] | type=2

    unsigned tensor_d1 = 0x100000u;                     // generous row count (no OOB)
    i32x8 g1;
    g1[0] = (int)((1u << 16) |                          // data_size = 2 bytes
                  (1u << 20) |                          // pad_enable
                  (pad_interval << 22) | (pad_amount << 25));
    g1[1] = (int)((row_stride & 0xFFFFu) << 16);        // tensor_dim0[15:0] @ bit48
    g1[2] = (int)(((row_stride >> 16) & 0xFFFFu) |      // tensor_dim0[31:16]
                  ((tensor_d1 & 0xFFFFu) << 16));       // tensor_dim1[15:0]
    g1[3] = (int)(((tensor_d1 >> 16) & 0xFFFFu) |       // tensor_dim1[31:16]
                  ((tile_k & 0xFFFFu) << 16));          // tile_dim0
    g1[4] = (int)(tile_rows & 0xFFFFu);                 // tile_dim1 (tile_dim2=0)
    g1[5] = (int)row_stride;                            // tensor_dim0_stride[31:0]
    g1[6] = 0;
    g1[7] = 0;

    i32x4 g2 = {0, 0, 0, 0};
    i32x4 g3 = {0, 0, 0, 0};
#if __clang_major__ >= 23
    i32x8 g4 = {0, 0, 0, 0, 0, 0, 0, 0};
    __builtin_amdgcn_tensor_load_to_lds(g0, g1, g2, g3, g4, 0);
#else
    __builtin_amdgcn_tensor_load_to_lds(g0, g1, g2, g3, 0);
#endif
}

// ---------------------------------------------------------------------------
// 16-lane cross-lane reductions, pure VALU DPP.
template <int CTRL>
__device__ __forceinline__ float dppf(float x) {
    return __int_as_float(
        __builtin_amdgcn_update_dpp(0, __float_as_int(x), CTRL, 0xF, 0xF, true));
}
__device__ __forceinline__ float red_max16(float x) {
    x = fmaxf(x, dppf<0xB1>(x));   // quad_perm xor1
    x = fmaxf(x, dppf<0x4E>(x));   // quad_perm xor2
    x = fmaxf(x, dppf<0x141>(x));  // row_half_mirror
    x = fmaxf(x, dppf<0x140>(x));  // row_mirror
    return x;
}
__device__ __forceinline__ float red_sum16(float x) {
    x += dppf<0xB1>(x);
    x += dppf<0x4E>(x);
    x += dppf<0x141>(x);
    x += dppf<0x140>(x);
    return x;
}

// ---------------------------------------------------------------------------
__global__ void cvt_f32_bf16(const float* __restrict__ in,
                             unsigned short* __restrict__ out, int n) {
    int i = blockIdx.x * blockDim.x + threadIdx.x;
    if (i < n) out[i] = f2bf(in[i]);
}

// ---------------------------------------------------------------------------
// C[M,N] = A[M,K]*W^T + bias; tiles staged by TDM, double-buffered.
// Block = 256 thr (8 waves), block tile 128x128; wave tile 32x64.
// MODE 0: bf16 row-major  MODE 1: bf16 V^T [b][h][d][s]  MODE 2: f32 row-major
#define GLD 40   // LDS row stride (32 + 8 pad) ushorts (= TDM pad 4 DWORDs / 64B)

template <int MODE>
__global__ void __launch_bounds__(256, 1)
gemm_bf16(const unsigned short* __restrict__ A,
          const unsigned short* __restrict__ W,
          const float* __restrict__ bias,
          void* __restrict__ out, int M, int N, int K) {
    __shared__ __align__(16) unsigned short sA[2][128 * GLD];
    __shared__ __align__(16) unsigned short sB[2][128 * GLD];

    int tid = threadIdx.x;
    int lane = tid & 31, wave = tid >> 5;
    int rowBlk = blockIdx.y * 128, colBlk = blockIdx.x * 128;
    int waveM = wave & 3, waveN = wave >> 2;      // 4 x 2 wave grid

    // Wave 0 issues two TDM descriptors per k-step: A 128x32 tile, B 128x32 tile.
    // Row = 32 bf16 = 64B = 8 x 8B units -> pad_interval 3; pad 4 DWORDs -> amount 3.
    auto stage = [&](int buf, int k) {
        if (wave == 0) {
            tdm_load_2d(A + (size_t)rowBlk * K + k, &sA[buf][0], 32, 128, K, 3, 3);
            tdm_load_2d(W + (size_t)colBlk * K + k, &sB[buf][0], 32, 128, K, 3, 3);
        }
    };
    auto publish = [&]() {
        if (wave == 0) __builtin_amdgcn_s_wait_tensorcnt(0);
        __syncthreads();
    };

    stage(0, 0);
    publish();

    v8f acc[2][4] = {};
    int buf = 0;
    for (int k = 0; k < K; k += 32) {
        if (k + 32 < K) stage(buf ^ 1, k + 32);   // overlap TDM with WMMAs

        v16bf af0 = load_frag_a(&sA[buf][0], GLD, waveM * 32,      0, lane);
        v16bf af1 = load_frag_a(&sA[buf][0], GLD, waveM * 32 + 16, 0, lane);
#pragma unroll
        for (int nt = 0; nt < 4; ++nt) {
            v16bf bfr = load_frag_b(&sB[buf][0], GLD, waveN * 64 + nt * 16, 0, lane);
            acc[0][nt] = wmma_bf16(af0, bfr, acc[0][nt]);
            acc[1][nt] = wmma_bf16(af1, bfr, acc[1][nt]);
        }
        publish();
        buf ^= 1;
    }

    int n = lane & 15, half = lane >> 4;
#pragma unroll
    for (int mt = 0; mt < 2; ++mt) {
#pragma unroll
        for (int nt = 0; nt < 4; ++nt) {
            int col = colBlk + waveN * 64 + nt * 16 + n;
            float bv = bias[col];
#pragma unroll
            for (int r = 0; r < 8; ++r) {
                int row = rowBlk + waveM * 32 + mt * 16 + r + half * 8;
                float val = acc[mt][nt][r] + bv;
                if (MODE == 0) {
                    ((unsigned short*)out)[(size_t)row * N + col] = f2bf(val);
                } else if (MODE == 1) {
                    int b = row >> 11, s = row & (SEQ - 1);
                    int h = col >> 7, d = col & (DHEAD - 1);
                    ((unsigned short*)out)[((size_t)(b * NHEADS + h) * DHEAD + d) * SEQ + s]
                        = f2bf(val);
                } else {
                    ((float*)out)[(size_t)row * N + col] = val;
                }
            }
        }
    }
}

// ---------------------------------------------------------------------------
// Flash attention, 64-key chunks; K/V tiles async-staged to LDS and shared by
// all 8 waves (each wave owns 16 query rows; block = 128 rows).
#define KLD 136   // K tile row stride (128 + 8) ushorts
#define VLD 72    // V^T tile row stride (64 + 8) ushorts

__global__ void __launch_bounds__(256, 1)
flash_attn(const unsigned short* __restrict__ Q,
           const unsigned short* __restrict__ Km,
           const unsigned short* __restrict__ Vt,
           unsigned short* __restrict__ Ctx) {
    __shared__ __align__(16) unsigned short sK[2][64 * KLD];    // 34 KB
    __shared__ __align__(16) unsigned short sV[2][128 * VLD];   // 36 KB
    __shared__ __align__(16) unsigned short ldsP[8][16 * 64];   // 16 KB

    int tid = threadIdx.x;
    int lane = tid & 31, wave = tid >> 5;
    int b = blockIdx.z, h = blockIdx.y;
    int q0 = blockIdx.x * 128 + wave * 16;
    int rowOff = b * SEQ;
    const unsigned short* vtB = Vt + (size_t)(b * NHEADS + h) * DHEAD * SEQ;

    auto stage = [&](int buf, int kc) {
#pragma unroll
        for (int t = 0; t < 4; ++t) {
            int c = tid + t * 256;                 // 1024 chunks of 16B each matrix
            int krow = c >> 4, kcol = c & 15;      // K: 64 rows x 16 chunks
            async_b128(Km + (size_t)(rowOff + kc + krow) * D_MODEL + h * DHEAD + kcol * 8,
                       &sK[buf][krow * KLD + kcol * 8]);
            int vrow = c >> 3, vcol = c & 7;       // V^T: 128 rows x 8 chunks
            async_b128(vtB + (size_t)vrow * SEQ + kc + vcol * 8,
                       &sV[buf][vrow * VLD + vcol * 8]);
        }
    };

    v16bf qf[4];
#pragma unroll
    for (int dc = 0; dc < 4; ++dc)
        qf[dc] = load_frag_a(Q, D_MODEL, rowOff + q0, h * DHEAD + dc * 32, lane);

    v8f acc[8] = {};
    float mrun[8], lrun[8];
#pragma unroll
    for (int r = 0; r < 8; ++r) { mrun[r] = -__builtin_inff(); lrun[r] = 0.f; }

    const float scale = 0.08838834764831845f;      // 1/sqrt(128)
    int n = lane & 15, half = lane >> 4;
    unsigned short* myP = ldsP[wave];

    stage(0, 0);
    wait_async0();
    __syncthreads();

    int buf = 0;
    for (int kc = 0; kc < SEQ; kc += 64) {
        if (kc + 64 < SEQ) stage(buf ^ 1, kc + 64);

        // scores: S = Q (16x128) x K^T (128x64) from LDS K tile
        v8f s[4] = {};
#pragma unroll
        for (int dc = 0; dc < 4; ++dc) {
#pragma unroll
            for (int j = 0; j < 4; ++j) {
                v16bf kb = load_frag_b(&sK[buf][0], KLD, j * 16, dc * 32, lane);
                s[j] = wmma_bf16(qf[dc], kb, s[j]);
            }
        }

        // online softmax per query row (row = r + half*8, key-lane = n)
#pragma unroll
        for (int r = 0; r < 8; ++r) {
            float a0 = s[0][r] * scale, a1 = s[1][r] * scale;
            float a2 = s[2][r] * scale, a3 = s[3][r] * scale;
            float mx = red_max16(fmaxf(fmaxf(a0, a1), fmaxf(a2, a3)));
            float mnew  = fmaxf(mrun[r], mx);
            float alpha = __expf(mrun[r] - mnew);
            float p0 = __expf(a0 - mnew), p1 = __expf(a1 - mnew);
            float p2 = __expf(a2 - mnew), p3 = __expf(a3 - mnew);
            float ps = red_sum16((p0 + p1) + (p2 + p3));
            lrun[r] = lrun[r] * alpha + ps;
            mrun[r] = mnew;
#pragma unroll
            for (int t = 0; t < 8; ++t) acc[t][r] *= alpha;
            int rowm = r + half * 8;
            myP[rowm * 64 +      n] = f2bf(p0);
            myP[rowm * 64 + 16 + n] = f2bf(p1);
            myP[rowm * 64 + 32 + n] = f2bf(p2);
            myP[rowm * 64 + 48 + n] = f2bf(p3);
        }

        // cross-lane relayout through per-wave LDS (DS in-order within wave)
        asm volatile("s_wait_dscnt 0x0" ::: "memory");
        __builtin_amdgcn_wave_barrier();

        const unsigned short* rp = myP + (lane & 15) * 64 + half * 8;
        FragBF pf0, pf1;
        pf0.q[0] = *(const uint4*)(rp);
        pf0.q[1] = *(const uint4*)(rp + 16);
        pf1.q[0] = *(const uint4*)(rp + 32);
        pf1.q[1] = *(const uint4*)(rp + 48);

#pragma unroll
        for (int t = 0; t < 8; ++t) {
            v16bf vb0 = load_frag_b(&sV[buf][0], VLD, t * 16, 0,  lane);
            acc[t] = wmma_bf16(pf0.v, vb0, acc[t]);
            v16bf vb1 = load_frag_b(&sV[buf][0], VLD, t * 16, 32, lane);
            acc[t] = wmma_bf16(pf1.v, vb1, acc[t]);
        }

        wait_async0();
        __syncthreads();
        buf ^= 1;
    }

    // epilogue: normalize, store context bf16
#pragma unroll
    for (int r = 0; r < 8; ++r) {
        float inv = 1.0f / lrun[r];
        int row = rowOff + q0 + r + half * 8;
#pragma unroll
        for (int t = 0; t < 8; ++t)
            Ctx[(size_t)row * D_MODEL + h * DHEAD + t * 16 + n] = f2bf(acc[t][r] * inv);
    }
}

// ---------------------------------------------------------------------------
extern "C" void kernel_launch(void* const* d_in, const int* in_sizes, int n_in,
                              void* d_out, int out_size, void* d_ws, size_t ws_size,
                              hipStream_t stream) {
    const float* x  = (const float*)d_in[0];
    const float* wq = (const float*)d_in[1];
    const float* bq = (const float*)d_in[2];
    const float* wk = (const float*)d_in[3];
    const float* bk = (const float*)d_in[4];
    const float* wv = (const float*)d_in[5];
    const float* bv = (const float*)d_in[6];
    const float* wo = (const float*)d_in[7];
    const float* bo = (const float*)d_in[8];

    const size_t XN = (size_t)MROWS * D_MODEL;
    const size_t WN = (size_t)D_MODEL * D_MODEL;

    char* p = (char*)d_ws;
    unsigned short* xb  = (unsigned short*)p; p += XN * 2;
    unsigned short* wqb = (unsigned short*)p; p += WN * 2;
    unsigned short* wkb = (unsigned short*)p; p += WN * 2;
    unsigned short* wvb = (unsigned short*)p; p += WN * 2;
    unsigned short* wob = (unsigned short*)p; p += WN * 2;
    unsigned short* Qb  = (unsigned short*)p; p += XN * 2;
    unsigned short* Kb  = (unsigned short*)p; p += XN * 2;
    unsigned short* Vtb = (unsigned short*)p; p += XN * 2;
    unsigned short* Ctx = (unsigned short*)p; p += XN * 2;

    cvt_f32_bf16<<<dim3((unsigned)((XN + 255) / 256)), 256, 0, stream>>>(x,  xb,  (int)XN);
    cvt_f32_bf16<<<dim3((unsigned)((WN + 255) / 256)), 256, 0, stream>>>(wq, wqb, (int)WN);
    cvt_f32_bf16<<<dim3((unsigned)((WN + 255) / 256)), 256, 0, stream>>>(wk, wkb, (int)WN);
    cvt_f32_bf16<<<dim3((unsigned)((WN + 255) / 256)), 256, 0, stream>>>(wv, wvb, (int)WN);
    cvt_f32_bf16<<<dim3((unsigned)((WN + 255) / 256)), 256, 0, stream>>>(wo, wob, (int)WN);

    dim3 ggrid(D_MODEL / 128, MROWS / 128);    // (16, 32)
    gemm_bf16<0><<<ggrid, 256, 0, stream>>>(xb, wqb, bq, Qb,  MROWS, D_MODEL, D_MODEL);
    gemm_bf16<0><<<ggrid, 256, 0, stream>>>(xb, wkb, bk, Kb,  MROWS, D_MODEL, D_MODEL);
    gemm_bf16<1><<<ggrid, 256, 0, stream>>>(xb, wvb, bv, Vtb, MROWS, D_MODEL, D_MODEL);

    flash_attn<<<dim3(SEQ / 128, NHEADS, BATCH), 256, 0, stream>>>(Qb, Kb, Vtb, Ctx);

    gemm_bf16<2><<<ggrid, 256, 0, stream>>>(Ctx, wob, bo, d_out, MROWS, D_MODEL, D_MODEL);
}
